// GAR_DSVDD_32049045963527
// MI455X (gfx1250) — matrix operand order repaired and checked by reading, not compiled, not run on hardware
//
#include <hip/hip_runtime.h>
#include <hip/hip_bf16.h>
#include <math.h>

typedef __attribute__((ext_vector_type(16))) __bf16 v16bf;
typedef __attribute__((ext_vector_type(8)))  float  v8f;

#define HEADS 8
#define DIM   64
#define DKK   32
#define KNN   16
#define GAMMA 0.5f
#define BETA  0.9f

union BfVec { v16bf v; uint4 q[2]; };

// Load two 16-byte chunks (8 bf16 each) into one v16bf WMMA operand.
template <typename PT>
static __device__ inline v16bf ld2(const PT* p0, const PT* p1) {
    BfVec u;
    u.q[0] = *(const uint4*)p0;
    u.q[1] = *(const uint4*)p1;
    return u.v;
}

// ---------- prep kernel 1: Z (f32) -> Z_bf16 ----------
__global__ void zcast_kernel(const float* __restrict__ z, __bf16* __restrict__ zb, int n) {
    int t = blockIdx.x * blockDim.x + threadIdx.x;
    if (t < n) zb[t] = (__bf16)z[t];
}

// ---------- prep kernel 2: M_cm[h][dp][d] = sum_k WQ[h][d][k] * WK[h][dp][k] ----------
__global__ void mker_kernel(const float* __restrict__ wq, const float* __restrict__ wk,
                            __bf16* __restrict__ mcm) {
    int t = blockIdx.x * blockDim.x + threadIdx.x;
    if (t >= HEADS * DIM * DIM) return;
    int h  = t >> 12;          // t / 4096
    int dp = (t >> 6) & 63;    // output-dim column
    int d  = t & 63;           // input-dim row (contiguous in M_cm)
    const float* q = wq + ((size_t)(h * DIM + d)  * DKK);
    const float* k = wk + ((size_t)(h * DIM + dp) * DKK);
    float acc = 0.f;
#pragma unroll
    for (int c = 0; c < DKK; ++c) acc += q[c] * k[c];
    mcm[t] = (__bf16)acc;
}

// ---------- main kernel: 1 wave32 per block, 16 nodes per block ----------
__global__ __launch_bounds__(32) void gar_main_kernel(
    const float* __restrict__ f, const int* __restrict__ idx,
    const float* __restrict__ ema, const __bf16* __restrict__ Zb,
    const __bf16* __restrict__ Mcm, float* __restrict__ out, int N)
{
    __shared__ __bf16 Vlds[16][16][DIM];   // [node][head(padded to 16)][dim], bf16
    __shared__ float  damp16[16];

    const int lane = threadIdx.x & 31;
    const int l16  = lane & 15;
    const int hi   = lane >> 4;            // 0 for lanes 0-15, 1 for 16-31
    const int i0   = blockIdx.x * 16;

    // Zero Vlds so padded head columns 8..15 contribute exact zeros.
    {
        uint4 z4 = {0u, 0u, 0u, 0u};
        uint4* p = (uint4*)&Vlds[0][0][0];
        const int total = (16 * 16 * DIM * 2) / 16;   // 2048 uint4
        for (int t = lane; t < total; t += 32) p[t] = z4;
    }
    __syncthreads();

    // ================= Stage A: V = Z_tile[16x64] @ M_h[64x64], all 8 heads ===========
    int rowA = i0 + l16; if (rowA >= N) rowA = N - 1;
    const __bf16* za = Zb + (size_t)rowA * DIM + hi * 8;
    v16bf a0 = ld2(za,      za + 16);       // K = 0..31  (A layout: 8-elem chunks)
    v16bf a1 = ld2(za + 32, za + 48);       // K = 32..63

    for (int h = 0; h < HEADS; ++h) {
        for (int nt = 0; nt < 4; ++nt) {
            v8f acc = {0.f, 0.f, 0.f, 0.f, 0.f, 0.f, 0.f, 0.f};
            // B operand: column (nt*16+l16) of M_h, 8-elem K chunks (same layout as A)
            const __bf16* mb = Mcm + ((size_t)(h * DIM) + nt * 16 + l16) * DIM + hi * 8;
            v16bf b0 = ld2(mb,      mb + 16);       // K = 0..31
            v16bf b1 = ld2(mb + 32, mb + 48);       // K = 32..63
            acc = __builtin_amdgcn_wmma_f32_16x16x32_bf16(false, a0, false, b0,
                                                          (short)0, acc, false, false);
            acc = __builtin_amdgcn_wmma_f32_16x16x32_bf16(false, a1, false, b1,
                                                          (short)0, acc, false, false);
            // C layout: VGPR r -> node (r + 8*hi), col -> dim nt*16+l16
            const int nh = hi * 8;
#pragma unroll
            for (int r = 0; r < 8; ++r)
                Vlds[nh + r][h][nt * 16 + l16] = (__bf16)acc[r];
        }
    }
    __syncthreads();

    // ================= Stage B: per node, logits[16 nei x 16 head] + softmax ==========
    const float scale = 0.17677669529663687f;  // 1/sqrt(32)
    for (int n = 0; n < 16; ++n) {
        int  i     = i0 + n;
        bool valid = (i < N);
        if (!valid) i = N - 1;

        int j = idx[(size_t)i * KNN + l16];
        if (j < 0) j = 0; if (j >= N) j = N - 1;

        float fi = f[i];
        float fj = f[j];
        if (lane < 16) damp16[lane] = GAMMA * (fmaxf(fi, 0.f) + fmaxf(fj, 0.f));
        __syncthreads();

        // A operand: gathered neighbor rows of Z_bf16 (L2-resident, 12.8 MB)
        const __bf16* zn = Zb + (size_t)j * DIM + hi * 8;
        v16bf an0 = ld2(zn,      zn + 16);
        v16bf an1 = ld2(zn + 32, zn + 48);
        // B operand: V_i^T columns = head rows of Vlds (zero-padded heads 8..15)
        const __bf16* vb = &Vlds[n][l16][hi * 8];
        v16bf bn0 = ld2(vb,      vb + 16);
        v16bf bn1 = ld2(vb + 32, vb + 48);

        v8f acc = {0.f, 0.f, 0.f, 0.f, 0.f, 0.f, 0.f, 0.f};
        acc = __builtin_amdgcn_wmma_f32_16x16x32_bf16(false, an0, false, bn0,
                                                      (short)0, acc, false, false);
        acc = __builtin_amdgcn_wmma_f32_16x16x32_bf16(false, an1, false, bn1,
                                                      (short)0, acc, false, false);

        // rows (neighbors) = r + 8*hi ; cols (heads) = l16
        float4 d0 = *(const float4*)&damp16[hi * 8];
        float4 d1 = *(const float4*)&damp16[hi * 8 + 4];
        float x[8];
        x[0] = acc[0] * scale - d0.x;  x[1] = acc[1] * scale - d0.y;
        x[2] = acc[2] * scale - d0.z;  x[3] = acc[3] * scale - d0.w;
        x[4] = acc[4] * scale - d1.x;  x[5] = acc[5] * scale - d1.y;
        x[6] = acc[6] * scale - d1.z;  x[7] = acc[7] * scale - d1.w;

        float m = x[0];
#pragma unroll
        for (int r = 1; r < 8; ++r) m = fmaxf(m, x[r]);
        m = fmaxf(m, __shfl_xor(m, 16, 32));      // combine rows 0-7 with 8-15

        float e[8], s = 0.f;
#pragma unroll
        for (int r = 0; r < 8; ++r) { e[r] = __expf(x[r] - m); s += e[r]; }
        s += __shfl_xor(s, 16, 32);
        float inv = 1.f / s;

        float w[8];
#pragma unroll
        for (int r = 0; r < 8; ++r) {
            float t = e[r] * inv;                  // softmax weight, head = l16
            t += __shfl_xor(t, 1, 32);             // sum over 8 head lanes
            t += __shfl_xor(t, 2, 32);
            t += __shfl_xor(t, 4, 32);
            w[r] = t * 0.125f;                     // mean over heads
        }

        if (valid && l16 == 0) {                   // lanes 0 (nei 0-7) and 16 (nei 8-15)
            const float* ep = ema + (size_t)i * KNN + hi * 8;
            float4 e0 = *(const float4*)ep;
            float4 e1 = *(const float4*)(ep + 4);
            float4 o0, o1;
            o0.x = fmaxf(BETA * e0.x + (1.f - BETA) * w[0], 0.f);
            o0.y = fmaxf(BETA * e0.y + (1.f - BETA) * w[1], 0.f);
            o0.z = fmaxf(BETA * e0.z + (1.f - BETA) * w[2], 0.f);
            o0.w = fmaxf(BETA * e0.w + (1.f - BETA) * w[3], 0.f);
            o1.x = fmaxf(BETA * e1.x + (1.f - BETA) * w[4], 0.f);
            o1.y = fmaxf(BETA * e1.y + (1.f - BETA) * w[5], 0.f);
            o1.z = fmaxf(BETA * e1.z + (1.f - BETA) * w[6], 0.f);
            o1.w = fmaxf(BETA * e1.w + (1.f - BETA) * w[7], 0.f);
            float* op = out + (size_t)i * KNN + hi * 8;
            *(float4*)op       = o0;
            *(float4*)(op + 4) = o1;
        }
        __syncthreads();
    }
}

extern "C" void kernel_launch(void* const* d_in, const int* in_sizes, int n_in,
                              void* d_out, int out_size, void* d_ws, size_t ws_size,
                              hipStream_t stream) {
    const float* Z   = (const float*)d_in[0];
    const float* f   = (const float*)d_in[1];
    const float* WQ  = (const float*)d_in[2];
    const float* WK  = (const float*)d_in[3];
    const float* ema = (const float*)d_in[4];
    const int*   idx = (const int*)d_in[5];
    float* out = (float*)d_out;
    const int N = in_sizes[1];                 // f has N elements

    __bf16* Zb  = (__bf16*)d_ws;               // N*64 bf16 = 12.8 MB (L2-resident)
    __bf16* Mcm = Zb + (size_t)N * DIM;        // 8*64*64 bf16 = 64 KB

    const int nz = N * DIM;
    zcast_kernel<<<(nz + 255) / 256, 256, 0, stream>>>(Z, Zb, nz);
    mker_kernel<<<(HEADS * DIM * DIM + 255) / 256, 256, 0, stream>>>(WQ, WK, Mcm);
    gar_main_kernel<<<(N + 15) / 16, 32, 0, stream>>>(f, idx, ema, Zb, Mcm, out, N);
}
